// HardNetLoss_1357209665920
// MI455X (gfx1250) — matrix operand order repaired and checked by reading, not compiled, not run on hardware
//
#include <hip/hip_runtime.h>
#include <cfloat>
#include <cstdint>
#include <cstddef>

// ---------------------------------------------------------------------------
// HardNet loss for MI455X (gfx1250).
//  * bf16 hi/lo (Markidis) split on the 16x16x32 bf16 WMMA pipe (~fp32 acc).
//  * Fused GEMM -> distance -> row/col min (distance matrices never hit HBM).
//  * Symmetric matrices (A.A^T, P.P^T): upper-triangle blocks only, each
//    off-diagonal tile feeds both row-min[i] and col-min[j]  (-33% FLOPs).
//  * Tile staging via the Tensor Data Mover (tensor_load_to_lds) with LDS
//    row padding done by the TDM descriptor itself; tracked with TENSORcnt.
// ---------------------------------------------------------------------------

#define BATCH      8192
#define DIM        128
#define MARGIN     0.2f
#define EPS_DIST   1e-6f
#define EPSF       1e-8f
#define DIAG_OFF   10.0f
#define MINTHRESH  0.008f

typedef __attribute__((ext_vector_type(16))) __bf16        bf16x16;
typedef __attribute__((ext_vector_type(8)))  float         f32x8;
typedef __attribute__((ext_vector_type(4)))  unsigned int  u32x4;
typedef __attribute__((ext_vector_type(8)))  int           i32x8;
typedef __attribute__((ext_vector_type(4)))  int           i32x4;

struct Pk { u32x4 a, b; };               // 32B carrier for a v16bf fragment

static constexpr int LDK = DIM + 8;      // LDS row stride in bf16 (272 B, 16B-aligned)

__device__ __forceinline__ unsigned short f2bf(float f) {   // RNE float->bf16
  unsigned int u = __float_as_uint(f);
  u += 0x7fffu + ((u >> 16) & 1u);
  return (unsigned short)(u >> 16);
}
__device__ __forceinline__ float bf2f(unsigned short h) {
  return __uint_as_float(((unsigned int)h) << 16);
}

// A-fragment (16x32 bf16) per ISA 7.12.2: lane%16 = row M; lanes<16 hold
// K = kk..kk+7 and kk+16..kk+23, lanes>=16 the +8 offsets.
__device__ __forceinline__ bf16x16 ldA(const unsigned short* s, int m, int kk) {
  Pk p;
  p.a = *(const u32x4*)(s + m * LDK + kk);
  p.b = *(const u32x4*)(s + m * LDK + kk + 16);
  return __builtin_bit_cast(bf16x16, p);
}
// B-fragment (32x16 bf16): lane%16 = col N; lanes<16 hold K = k0..k0+7 and
// k0+16..k0+23, lanes>=16 the +8 offsets (same order as A).
__device__ __forceinline__ bf16x16 ldB(const unsigned short* s, int n, int kk) {
  Pk p;
  p.a = *(const u32x4*)(s + n * LDK + kk);
  p.b = *(const u32x4*)(s + n * LDK + kk + 16);
  return __builtin_bit_cast(bf16x16, p);
}

// ---------------------------------------------------------------------------
// TDM: DMA one 128x128 bf16 tile (row stride 128 elems in global) into LDS,
// padding each 256B row to 272B (pad_interval=64 DWORDs -> enc 5,
// pad_amount=4 DWORDs -> enc 3).  D# per CDNA5 ISA ch.8.
// ---------------------------------------------------------------------------
__device__ __forceinline__ void tdm_load_tile(const unsigned short* gsrc,
                                              unsigned ldsByteAddr) {
  unsigned long long ga = (unsigned long long)(uintptr_t)gsrc;
  u32x4 g0;
  g0.x = 1u;                                            // count=1, user mode
  g0.y = ldsByteAddr;                                   // lds_addr[31:0]
  g0.z = (unsigned)(ga & 0xFFFFFFFFull);                // global_addr[31:0]
  g0.w = (unsigned)((ga >> 32) & 0x1FFFFFFull)          // global_addr[56:32]
       | (2u << 30);                                    // type=2 ("image")
  i32x8 g1;
  g1[0] = (int)((1u << 16)        // data_size = 1 -> 2 bytes/elem
              | (1u << 20)        // pad_enable
              | (5u << 22)        // pad_interval: 64 DWORDs
              | (3u << 25));      // pad_amount:   4 DWORDs
  g1[1] = (int)(128u << 16);      // tensor_dim0[15:0] = 128
  g1[2] = (int)(128u << 16);      // tensor_dim1[15:0] = 128
  g1[3] = (int)(128u << 16);      // tile_dim0 = 128
  g1[4] = (int)128u;              // tile_dim1 = 128, tile_dim2 = 0
  g1[5] = (int)128u;              // tensor_dim0_stride[31:0] = 128
  g1[6] = 0;
  g1[7] = 0;
  i32x4 gz = {0, 0, 0, 0};        // groups 2/3 unused for 2D tiles
#if __clang_major__ >= 23
  i32x8 gz8 = {0, 0, 0, 0, 0, 0, 0, 0};
  __builtin_amdgcn_tensor_load_to_lds(g0, g1, gz, gz, gz8, 0);
#else
  __builtin_amdgcn_tensor_load_to_lds(g0, g1, gz, gz, 0);
#endif
}

// ---------------------------------------------------------------------------
// 1) Deinterleave x into anchors/positives, bf16 hi/lo split + fp32 sq-norms.
// ---------------------------------------------------------------------------
__global__ __launch_bounds__(256) void prep_kernel(
    const float* __restrict__ x,
    unsigned short* __restrict__ Ahi, unsigned short* __restrict__ Alo,
    unsigned short* __restrict__ Phi, unsigned short* __restrict__ Plo,
    float* __restrict__ asq, float* __restrict__ psq)
{
  int wave = (blockIdx.x * blockDim.x + threadIdx.x) >> 5;
  int lane = threadIdx.x & 31;
  if (wave >= BATCH) return;
  const float* ar = x + (size_t)(3 * wave) * DIM;       // x[0::3]
  const float* pr = ar + DIM;                           // x[1::3]
  float sa = 0.f, sp = 0.f;
  for (int c = lane; c < DIM; c += 32) {
    float fa = ar[c];
    unsigned short ha = f2bf(fa);
    Ahi[(size_t)wave * DIM + c] = ha;
    Alo[(size_t)wave * DIM + c] = f2bf(fa - bf2f(ha));
    sa += fa * fa;
    float fp = pr[c];
    unsigned short hp = f2bf(fp);
    Phi[(size_t)wave * DIM + c] = hp;
    Plo[(size_t)wave * DIM + c] = f2bf(fp - bf2f(hp));
    sp += fp * fp;
  }
  for (int o = 16; o > 0; o >>= 1) {
    sa += __shfl_xor(sa, o, 32);
    sp += __shfl_xor(sp, o, 32);
  }
  if (lane == 0) { asq[wave] = sa; psq[wave] = sp; }
}

// ---------------------------------------------------------------------------
// 2) Init the three row-min arrays (contiguous) to +FLT_MAX every launch.
// ---------------------------------------------------------------------------
__global__ void init_min_kernel(float* __restrict__ m, int n) {
  int i = blockIdx.x * blockDim.x + threadIdx.x;
  if (i < n) m[i] = FLT_MAX;
}

// ---------------------------------------------------------------------------
// 3) Fused GEMM(bf16 hi/lo WMMA) -> distance -> row/col-min.
//    z: 0 = d(A,P) full grid;  1 = d(A,A), 2 = d(P,P): upper triangle only.
// ---------------------------------------------------------------------------
__global__ __launch_bounds__(256) void dist_min_kernel(
    const unsigned short* __restrict__ Ahi, const unsigned short* __restrict__ Alo,
    const unsigned short* __restrict__ Phi, const unsigned short* __restrict__ Plo,
    const float* __restrict__ asq, const float* __restrict__ psq,
    float* __restrict__ minAP, float* __restrict__ minAA, float* __restrict__ minPP,
    float* __restrict__ posOut)
{
  const int mode = blockIdx.z;
  const bool sym = (mode != 0);
  if (sym && blockIdx.x < blockIdx.y) return;          // symmetric: skip lower
  const bool colAlso = sym && (blockIdx.x != blockIdx.y);

  extern __shared__ char smem[];
  unsigned short* sXhi = (unsigned short*)smem;
  unsigned short* sXlo = sXhi + 128 * LDK;
  unsigned short* sYhi = sXlo + 128 * LDK;
  unsigned short* sYlo = sYhi + 128 * LDK;
  float* sXsq = (float*)(sYlo + 128 * LDK);
  float* sYsq = sXsq + 128;

  const unsigned short *Xhi, *Xlo, *Yhi, *Ylo;
  const float *xs, *ys;
  float* mArr;
  if (mode == 0)      { Xhi = Ahi; Xlo = Alo; Yhi = Phi; Ylo = Plo; xs = asq; ys = psq; mArr = minAP; }
  else if (mode == 1) { Xhi = Ahi; Xlo = Alo; Yhi = Ahi; Ylo = Alo; xs = asq; ys = asq; mArr = minAA; }
  else                { Xhi = Phi; Xlo = Plo; Yhi = Phi; Ylo = Plo; xs = psq; ys = psq; mArr = minPP; }

  const int iTile = blockIdx.y * 128;   // X rows (output rows)
  const int jTile = blockIdx.x * 128;   // Y rows (output cols)
  const int t = threadIdx.x;

  // ---- stage 4 tiles via the Tensor Data Mover (wave 0 issues, all wait) ---
  if (t < 32) {
    tdm_load_tile(Xhi + (size_t)iTile * DIM, (unsigned)(uintptr_t)sXhi);
    tdm_load_tile(Xlo + (size_t)iTile * DIM, (unsigned)(uintptr_t)sXlo);
    tdm_load_tile(Yhi + (size_t)jTile * DIM, (unsigned)(uintptr_t)sYhi);
    tdm_load_tile(Ylo + (size_t)jTile * DIM, (unsigned)(uintptr_t)sYlo);
    __builtin_amdgcn_s_wait_tensorcnt(0);
  }
  if (t < 128)       sXsq[t]       = xs[iTile + t];
  else               sYsq[t - 128] = ys[jTile + (t - 128)];
  __syncthreads();

  // ---- wave tiling: 8 waves as 4(row)x2(col); wave = 32x64 sub-tile ----
  const int wave = t >> 5, lane = t & 31;
  const int wy = wave & 3, wx = wave >> 2;
  const int rowBase = wy * 32;
  const int colBase = wx * 64;
  const int half = lane >> 4;     // C/D: lanes>=16 hold rows M+8
  const int l16 = lane & 15;

  f32x8 acc[2][4] = {};

#pragma unroll
  for (int k0 = 0; k0 < DIM; k0 += 32) {
    bf16x16 aHi[2], aLo[2];
#pragma unroll
    for (int tr = 0; tr < 2; ++tr) {
      int m  = rowBase + tr * 16 + l16;
      int kk = k0 + half * 8;
      aHi[tr] = ldA(sXhi, m, kk);
      aLo[tr] = ldA(sXlo, m, kk);
    }
#pragma unroll
    for (int tc = 0; tc < 4; ++tc) {
      int n  = colBase + tc * 16 + l16;
      int kk = k0 + half * 8;
      bf16x16 bHi = ldB(sYhi, n, kk);
      bf16x16 bLo = ldB(sYlo, n, kk);
#pragma unroll
      for (int tr = 0; tr < 2; ++tr) {
        // dot ~= lo*hi + hi*lo + hi*hi  (Markidis split, lo*lo dropped)
        acc[tr][tc] = __builtin_amdgcn_wmma_f32_16x16x32_bf16(
            false, aLo[tr], false, bHi, (short)0, acc[tr][tc], false, false);
        acc[tr][tc] = __builtin_amdgcn_wmma_f32_16x16x32_bf16(
            false, aHi[tr], false, bLo, (short)0, acc[tr][tc], false, false);
        acc[tr][tc] = __builtin_amdgcn_wmma_f32_16x16x32_bf16(
            false, aHi[tr], false, bHi, (short)0, acc[tr][tc], false, false);
      }
    }
  }

  // ---- epilogue: dot -> distance, diag/threshold, row (+col) mins ----
  const bool m0 = (mode == 0);
  float rmin[2][8];
  float cmin[4];
#pragma unroll
  for (int tr = 0; tr < 2; ++tr)
#pragma unroll
    for (int v = 0; v < 8; ++v) rmin[tr][v] = FLT_MAX;
#pragma unroll
  for (int tc = 0; tc < 4; ++tc) cmin[tc] = FLT_MAX;

#pragma unroll
  for (int tr = 0; tr < 2; ++tr) {
#pragma unroll
    for (int tc = 0; tc < 4; ++tc) {
#pragma unroll
      for (int v = 0; v < 8; ++v) {
        int il = rowBase + tr * 16 + half * 8 + v;   // C layout: VGPR v -> row
        int jl = colBase + tc * 16 + l16;            // lane%16 -> col
        float dot  = acc[tr][tc][v];
        float val  = sXsq[il] + sYsq[jl] - 2.0f * dot + EPS_DIST;
        float base = sqrtf(fmaxf(val, 0.0f)) + EPSF; // = d_* + EPS
        int gi = iTile + il, gj = jTile + jl;
        float dd = base;
        if (gi == gj) {
          if (m0) posOut[gi] = base;                 // pos = diag(d_ap)
          dd += DIAG_OFF;
        }
        if (m0 && dd < MINTHRESH) dd += DIAG_OFF;
        rmin[tr][v] = fminf(rmin[tr][v], dd);
        if (colAlso) cmin[tc] = fminf(cmin[tc], dd); // symmetric transpose min
      }
    }
  }

#pragma unroll
  for (int tr = 0; tr < 2; ++tr) {
#pragma unroll
    for (int v = 0; v < 8; ++v) {
      float r = rmin[tr][v];
      r = fminf(r, __shfl_xor(r, 1, 32));
      r = fminf(r, __shfl_xor(r, 2, 32));
      r = fminf(r, __shfl_xor(r, 4, 32));
      r = fminf(r, __shfl_xor(r, 8, 32));
      if (l16 == 0) {
        int gi = iTile + rowBase + tr * 16 + half * 8 + v;
        // all distances > 0 => IEEE order == int order of the bit patterns
        atomicMin((int*)mArr + gi, __float_as_int(r));
      }
    }
  }
  if (colAlso) {
#pragma unroll
    for (int tc = 0; tc < 4; ++tc) {
      float c = cmin[tc];
      c = fminf(c, __shfl_xor(c, 16, 32));           // merge row-halves
      if (half == 0) {
        int gj = jTile + colBase + tc * 16 + l16;
        atomicMin((int*)mArr + gj, __float_as_int(c));
      }
    }
  }
}

// ---------------------------------------------------------------------------
// 4) min_neg = min of the three mins; loss = mean(relu(margin + pos - min)).
// ---------------------------------------------------------------------------
__global__ __launch_bounds__(256) void finalize_kernel(
    float* __restrict__ out,
    const float* __restrict__ minAP, const float* __restrict__ minAA,
    const float* __restrict__ minPP)
{
  __shared__ float red[256];
  int t = threadIdx.x;
  float s = 0.0f;
  for (int i = t; i < BATCH; i += 256) {
    float mn = fminf(minAP[i], fminf(minAA[i], minPP[i]));
    out[BATCH + i] = mn;
    s += fmaxf(MARGIN + out[i] - mn, 0.0f);
  }
  red[t] = s;
  __syncthreads();
  for (int o = 128; o > 0; o >>= 1) {
    if (t < o) red[t] += red[t + o];
    __syncthreads();
  }
  if (t == 0) out[2 * BATCH] = red[0] / (float)BATCH;
}

// ---------------------------------------------------------------------------
extern "C" void kernel_launch(void* const* d_in, const int* in_sizes, int n_in,
                              void* d_out, int out_size, void* d_ws, size_t ws_size,
                              hipStream_t stream) {
  (void)in_sizes; (void)n_in; (void)out_size; (void)ws_size;
  const float* x = (const float*)d_in[0];
  float* out = (float*)d_out;

  // workspace carve (~8.2 MB total)
  char* w = (char*)d_ws;
  const size_t nBF = (size_t)BATCH * DIM * sizeof(unsigned short); // 2 MiB each
  unsigned short* Ahi = (unsigned short*)w; w += nBF;
  unsigned short* Alo = (unsigned short*)w; w += nBF;
  unsigned short* Phi = (unsigned short*)w; w += nBF;
  unsigned short* Plo = (unsigned short*)w; w += nBF;
  float* asq   = (float*)w; w += BATCH * sizeof(float);
  float* psq   = (float*)w; w += BATCH * sizeof(float);
  float* minAP = (float*)w; w += BATCH * sizeof(float);
  float* minAA = (float*)w; w += BATCH * sizeof(float);
  float* minPP = (float*)w; w += BATCH * sizeof(float);

  prep_kernel<<<BATCH / 8, 256, 0, stream>>>(x, Ahi, Alo, Phi, Plo, asq, psq);
  init_min_kernel<<<(3 * BATCH) / 256, 256, 0, stream>>>(minAP, 3 * BATCH);

  const size_t smemBytes = 4u * 128u * LDK * sizeof(unsigned short)
                         + 2u * 128u * sizeof(float);              // ~137 KB
  dim3 grid(BATCH / 128, BATCH / 128, 3);
  dist_min_kernel<<<grid, 256, smemBytes, stream>>>(
      Ahi, Alo, Phi, Plo, asq, psq, minAP, minAA, minPP, out);

  finalize_kernel<<<1, 256, 0, stream>>>(out, minAP, minAA, minPP);
}
